// _MHA_38500086841667
// MI455X (gfx1250) — matrix operand order repaired and checked by reading, not compiled, not run on hardware
//
#include <hip/hip_runtime.h>
#include <hip/hip_bf16.h>

// MHA forward for MI455X (gfx1250): bf16 WMMA (v_wmma_f32_16x16x32_bf16) with
// f32 accumulation, flash-style streaming softmax, bf16 intermediates in d_ws.
// Round 4: scalarize Q/K/V destination select in qkv epilogue (readfirstlane)
// to kill the per-element exec-mask dance seen in round-3 asm.

typedef __attribute__((ext_vector_type(16))) __bf16 bf16x16;
typedef __attribute__((ext_vector_type(8)))  float  f32x8;
typedef __attribute__((ext_vector_type(4)))  int    i32x4;

union FragU { bf16x16 v; unsigned u[8]; };

// 16-bit A-matrix 16x32: VGPR v, lane-half h -> k pair base
__device__ __forceinline__ int ka_base(int v, int half) {
    return ((v & 4) << 2) + half * 8 + ((v & 3) << 1);
}
// 16-bit B-matrix 32x16: same K order as A (VGPR v, lane-half h -> k pair base)
__device__ __forceinline__ int kb_base(int v, int half) {
    return ((v & 4) << 2) + half * 8 + ((v & 3) << 1);
}

// Fragment from an LDS tile laid out [outer][k], row stride RS bf16 elements
// (RS even): 8 aligned dword loads -> compiler merges into ds_load_b128 pairs.
template <int RS, bool AFRAG>
__device__ __forceinline__ bf16x16 load_frag(const __hip_bfloat16* base,
                                             int outer, int kofs, int lane) {
    FragU f;
    const int half = lane >> 4;
    const unsigned short* p =
        reinterpret_cast<const unsigned short*>(base) + (size_t)outer * RS;
#pragma unroll
    for (int v = 0; v < 8; ++v) {
        int k0 = kofs + (AFRAG ? ka_base(v, half) : kb_base(v, half));
        f.u[v] = *reinterpret_cast<const unsigned*>(p + k0);
    }
    return f.v;
}

__device__ __forceinline__ unsigned pack_bf16(float lo, float hi) {
    union { __hip_bfloat16 h[2]; unsigned u; } p;
    p.h[0] = __float2bfloat16(lo);
    p.h[1] = __float2bfloat16(hi);
    return p.u;
}

#define WMMA_BF16(a, b, c) \
    __builtin_amdgcn_wmma_f32_16x16x32_bf16(false, (a), false, (b), (short)0, (c), false, false)

// Async 16B global->LDS copy (CDNA5 GLOBAL_LOAD_ASYNC_TO_LDS_B128), with
// synchronous uint4 fallback if the builtin is unavailable.
#if __has_builtin(__builtin_amdgcn_global_load_async_to_lds_b128)
#define ASYNC_CP16(gp, lp)                                                     \
    __builtin_amdgcn_global_load_async_to_lds_b128(                            \
        (__attribute__((address_space(1))) i32x4*)(gp),                        \
        (__attribute__((address_space(3))) i32x4*)(lp), 0, 0)
#if __has_builtin(__builtin_amdgcn_s_wait_asynccnt)
#define ASYNC_WAIT() __builtin_amdgcn_s_wait_asynccnt(0)
#else
#define ASYNC_WAIT() asm volatile("s_wait_asynccnt 0" ::: "memory")
#endif
#else
#define ASYNC_CP16(gp, lp) \
    (*reinterpret_cast<uint4*>(lp) = *reinterpret_cast<const uint4*>(gp))
#define ASYNC_WAIT() ((void)0)
#endif

// ---------------------------------------------------------------------------
// Kernel 1: qkv = x @ W_qkv + b_qkv, scattered into Q (x 1/8), K, V [B,H,T,d] bf16
// grid (64, 24), block 256 (8 waves: wm 0..3, wn 0..1).  M=8192, N=3072, K=1024.
// Block tile 128x128, wave tile 32x64, k-chunk 32.
// ---------------------------------------------------------------------------
__global__ __launch_bounds__(256) void qkv_gemm(
    const float* __restrict__ x, const float* __restrict__ Wqkv,
    const float* __restrict__ bqkv,
    __hip_bfloat16* __restrict__ qd, __hip_bfloat16* __restrict__ kd,
    __hip_bfloat16* __restrict__ vd) {
    __shared__ __align__(16) __hip_bfloat16 Al[128][40];  // [m][k]
    __shared__ __align__(16) __hip_bfloat16 Bl[128][40];  // [n][k]
    const int tid = threadIdx.x, lane = tid & 31, w = tid >> 5;
    const int wm = w & 3, wn = w >> 2;
    const int mb = blockIdx.x * 128, nb = blockIdx.y * 128;
    const int half = lane >> 4, ln = lane & 15;
    f32x8 acc[2][4] = {};

    // Staging coordinates (compile-time-friendly).
    const int am = tid >> 3;          // A: row group 0..31 (+32 per j)
    const int akq = (tid & 7) * 4;    // A: k offset (float4)
    const int bn = (tid & 31) * 4;    // B: n offset (float4)
    const int bk = (tid >> 5) * 2;    // B: k-pair 0..14 (+16 per j)

    for (int kb = 0; kb < 1024; kb += 32) {
        __syncthreads();
#pragma unroll
        for (int j = 0; j < 4; ++j) {                 // A: 128x32 f32 -> bf16
            int m = am + j * 32;
            float4 f = *reinterpret_cast<const float4*>(
                &x[(size_t)(mb + m) * 1024 + kb + akq]);
            union { unsigned u[2]; uint2 u2; } p;
            p.u[0] = pack_bf16(f.x, f.y);
            p.u[1] = pack_bf16(f.z, f.w);
            *reinterpret_cast<uint2*>(&Al[m][akq]) = p.u2;
        }
#pragma unroll
        for (int j = 0; j < 2; ++j) {                 // B: 32x128 f32 -> bf16 [n][k]
            int kk = bk + j * 16;
            const float* wp = &Wqkv[(size_t)(kb + kk) * 3072 + nb + bn];
            float4 r0 = *reinterpret_cast<const float4*>(wp);
            float4 r1 = *reinterpret_cast<const float4*>(wp + 3072);
            *reinterpret_cast<unsigned*>(&Bl[bn + 0][kk]) = pack_bf16(r0.x, r1.x);
            *reinterpret_cast<unsigned*>(&Bl[bn + 1][kk]) = pack_bf16(r0.y, r1.y);
            *reinterpret_cast<unsigned*>(&Bl[bn + 2][kk]) = pack_bf16(r0.z, r1.z);
            *reinterpret_cast<unsigned*>(&Bl[bn + 3][kk]) = pack_bf16(r0.w, r1.w);
        }
        __syncthreads();
        bf16x16 af[2], bfg[4];
#pragma unroll
        for (int i = 0; i < 2; ++i)
            af[i] = load_frag<40, true>(&Al[0][0], wm * 32 + i * 16 + ln, 0, lane);
#pragma unroll
        for (int j = 0; j < 4; ++j)
            bfg[j] = load_frag<40, false>(&Bl[0][0], wn * 64 + j * 16 + ln, 0, lane);
#pragma unroll
        for (int i = 0; i < 2; ++i)
#pragma unroll
            for (int j = 0; j < 4; ++j)
                acc[i][j] = WMMA_BF16(af[i], bfg[j], acc[i][j]);
    }

    // Epilogue: each 16-column fragment lies entirely inside one of Q/K/V
    // (16-aligned blocks never straddle a 1024 boundary), so the destination
    // select is wave-uniform -> force it scalar with readfirstlane.
#pragma unroll
    for (int j = 0; j < 4; ++j) {
        const int ncol0 = __builtin_amdgcn_readfirstlane(nb + wn * 64 + j * 16);
        const int which = ncol0 >> 10;            // 0=q, 1=k, 2=v (scalar)
        const int cc0 = ncol0 & 1023;             // col within the 1024 block
        const int h = cc0 >> 6, dd0 = cc0 & 63;   // head, d offset (scalar)
        __hip_bfloat16* dst = (which == 0) ? qd : ((which == 1) ? kd : vd);
        const float scale = (which == 0) ? 0.125f : 1.0f;
        const float bias = bqkv[ncol0 + ln];
#pragma unroll
        for (int i = 0; i < 2; ++i) {
#pragma unroll
            for (int vi = 0; vi < 8; ++vi) {
                int m = mb + wm * 32 + i * 16 + vi + half * 8;
                int bb = m >> 11, t = m & 2047;
                size_t di = ((size_t)(bb * 16 + h) * 2048 + t) * 64 + dd0 + ln;
                dst[di] = __float2bfloat16((acc[i][j][vi] + bias) * scale);
            }
        }
    }
}

// ---------------------------------------------------------------------------
// Kernel 2: flash attention per (b,h). grid (64, 32), block 128 (4 waves).
// Each wave owns 16 query rows; key blocks of 64; O written as [B,T,H*d] bf16.
// ---------------------------------------------------------------------------
__global__ __launch_bounds__(128) void attn(
    const __hip_bfloat16* __restrict__ qd, const __hip_bfloat16* __restrict__ kd,
    const __hip_bfloat16* __restrict__ vd, __hip_bfloat16* __restrict__ od) {
    __shared__ __align__(16) __hip_bfloat16 Kt[64][72];     // [key][d]
    __shared__ __align__(16) __hip_bfloat16 Vt[64][72];     // [d][key] (transposed)
    __shared__ __align__(16) __hip_bfloat16 Pl[4][16][72];  // per-wave P scratch [m][key]
    const int tid = threadIdx.x, lane = tid & 31, w = tid >> 5;
    const int bh = blockIdx.x, qb = blockIdx.y * 64;
    const int half = lane >> 4, ln = lane & 15;

    // Q fragments (A layout) straight from global, kept in registers.
    bf16x16 qf[2];
    {
        const unsigned short* qrow = reinterpret_cast<const unsigned short*>(
            qd + ((size_t)bh * 2048 + qb + w * 16 + ln) * 64);
#pragma unroll
        for (int kc = 0; kc < 2; ++kc) {
            FragU f;
#pragma unroll
            for (int v = 0; v < 8; ++v) {
                int k0 = kc * 32 + ka_base(v, half);
                f.u[v] = *reinterpret_cast<const unsigned*>(qrow + k0);
            }
            qf[kc] = f.v;
        }
    }

    float m_run[8], l_run[8];
#pragma unroll
    for (int vi = 0; vi < 8; ++vi) { m_run[vi] = -1e30f; l_run[vi] = 0.f; }
    f32x8 oa[4] = {};

    const __hip_bfloat16* kbase = kd + (size_t)bh * 2048 * 64;
    const __hip_bfloat16* vbase = vd + (size_t)bh * 2048 * 64;
    const int sr = tid >> 3, sc8 = (tid & 7) * 8;   // staging coords (+16 rows / j)

    for (int kblk = 0; kblk < 2048; kblk += 64) {
        __syncthreads();
#pragma unroll
        for (int j = 0; j < 4; ++j) {
            int r = sr + j * 16;
            ASYNC_CP16(kbase + (size_t)(kblk + r) * 64 + sc8, &Kt[r][sc8]);
            union { uint4 q4; __hip_bfloat16 h[8]; } vv;
            vv.q4 = *reinterpret_cast<const uint4*>(vbase + (size_t)(kblk + r) * 64 + sc8);
#pragma unroll
            for (int jj = 0; jj < 8; ++jj) Vt[sc8 + jj][r] = vv.h[jj];
        }
        ASYNC_WAIT();
        __syncthreads();

        // S = Q @ K^T  (B[k=d][n=key] = Kt[key][d])
        f32x8 s[4] = {};
#pragma unroll
        for (int nf = 0; nf < 4; ++nf)
#pragma unroll
            for (int kc = 0; kc < 2; ++kc) {
                bf16x16 bk = load_frag<72, false>(&Kt[0][0], nf * 16 + ln, kc * 32, lane);
                s[nf] = WMMA_BF16(qf[kc], bk, s[nf]);
            }

        // Online softmax: row = vi + 8*half lives within one 16-lane half.
        float mp[8], sp[8];
#pragma unroll
        for (int vi = 0; vi < 8; ++vi)
            mp[vi] = fmaxf(fmaxf(s[0][vi], s[1][vi]), fmaxf(s[2][vi], s[3][vi]));
#pragma unroll
        for (int mask = 1; mask < 16; mask <<= 1)
#pragma unroll
            for (int vi = 0; vi < 8; ++vi)
                mp[vi] = fmaxf(mp[vi], __shfl_xor(mp[vi], mask, 32));
#pragma unroll
        for (int vi = 0; vi < 8; ++vi) {
            float mnew = fmaxf(m_run[vi], mp[vi]);
            float alpha = __expf(m_run[vi] - mnew);
            m_run[vi] = mnew;
            float acc = 0.f;
#pragma unroll
            for (int nf = 0; nf < 4; ++nf) {
                float p = __expf(s[nf][vi] - mnew);
                s[nf][vi] = p;
                acc += p;
            }
            sp[vi] = acc;
            l_run[vi] = l_run[vi] * alpha;
#pragma unroll
            for (int nf = 0; nf < 4; ++nf) oa[nf][vi] = oa[nf][vi] * alpha;
        }
#pragma unroll
        for (int mask = 1; mask < 16; mask <<= 1)
#pragma unroll
            for (int vi = 0; vi < 8; ++vi) sp[vi] += __shfl_xor(sp[vi], mask, 32);
#pragma unroll
        for (int vi = 0; vi < 8; ++vi) l_run[vi] += sp[vi];

        // C-layout P -> per-wave LDS scratch (bf16) -> A-layout fragments.
#pragma unroll
        for (int nf = 0; nf < 4; ++nf)
#pragma unroll
            for (int vi = 0; vi < 8; ++vi)
                Pl[w][vi + half * 8][nf * 16 + ln] = __float2bfloat16(s[nf][vi]);

        bf16x16 pa[2];
#pragma unroll
        for (int kc = 0; kc < 2; ++kc)
            pa[kc] = load_frag<72, true>(&Pl[w][0][0], ln, kc * 32, lane);
#pragma unroll
        for (int nf = 0; nf < 4; ++nf)
#pragma unroll
            for (int kc = 0; kc < 2; ++kc) {
                bf16x16 vf = load_frag<72, false>(&Vt[0][0], nf * 16 + ln, kc * 32, lane);
                oa[nf] = WMMA_BF16(pa[kc], vf, oa[nf]);
            }
    }

    // Normalize and store O as [B, T, H*d].
    const int b = bh >> 4, h = bh & 15;
#pragma unroll
    for (int vi = 0; vi < 8; ++vi) {
        float inv = 1.f / l_run[vi];
        int r = qb + w * 16 + vi + half * 8;
        size_t row = ((size_t)b * 2048 + r) * 1024 + h * 64;
#pragma unroll
        for (int nf = 0; nf < 4; ++nf)
            od[row + nf * 16 + ln] = __float2bfloat16(oa[nf][vi] * inv);
    }
}

// ---------------------------------------------------------------------------
// Kernel 3: y = o @ W_out + b_out (f32 out). grid (64, 8), block 256.
// Block tile 128x128, wave tile 32x64.
// ---------------------------------------------------------------------------
__global__ __launch_bounds__(256) void out_gemm(
    const __hip_bfloat16* __restrict__ od, const float* __restrict__ Wout,
    const float* __restrict__ bout, float* __restrict__ y) {
    __shared__ __align__(16) __hip_bfloat16 Al[128][40];
    __shared__ __align__(16) __hip_bfloat16 Bl[128][40];
    const int tid = threadIdx.x, lane = tid & 31, w = tid >> 5;
    const int wm = w & 3, wn = w >> 2;
    const int mb = blockIdx.x * 128, nb = blockIdx.y * 128;
    const int half = lane >> 4, ln = lane & 15;
    f32x8 acc[2][4] = {};

    const int am = tid >> 2;          // A: row 0..63 (+64 per j), uint4 copy
    const int akq8 = (tid & 3) * 8;   // A: k offset (8 bf16)
    const int bn = (tid & 31) * 4;
    const int bk = (tid >> 5) * 2;

    for (int kb = 0; kb < 1024; kb += 32) {
        __syncthreads();
#pragma unroll
        for (int j = 0; j < 2; ++j) {                 // A: 128x32 bf16 copy
            int m = am + j * 64;
            ASYNC_CP16(&od[(size_t)(mb + m) * 1024 + kb + akq8], &Al[m][akq8]);
        }
#pragma unroll
        for (int j = 0; j < 2; ++j) {                 // B: 32x128 f32 -> bf16 [n][k]
            int kk = bk + j * 16;
            const float* wp = &Wout[(size_t)(kb + kk) * 1024 + nb + bn];
            float4 r0 = *reinterpret_cast<const float4*>(wp);
            float4 r1 = *reinterpret_cast<const float4*>(wp + 1024);
            *reinterpret_cast<unsigned*>(&Bl[bn + 0][kk]) = pack_bf16(r0.x, r1.x);
            *reinterpret_cast<unsigned*>(&Bl[bn + 1][kk]) = pack_bf16(r0.y, r1.y);
            *reinterpret_cast<unsigned*>(&Bl[bn + 2][kk]) = pack_bf16(r0.z, r1.z);
            *reinterpret_cast<unsigned*>(&Bl[bn + 3][kk]) = pack_bf16(r0.w, r1.w);
        }
        ASYNC_WAIT();
        __syncthreads();
        bf16x16 af[2], bfg[4];
#pragma unroll
        for (int i = 0; i < 2; ++i)
            af[i] = load_frag<40, true>(&Al[0][0], wm * 32 + i * 16 + ln, 0, lane);
#pragma unroll
        for (int j = 0; j < 4; ++j)
            bfg[j] = load_frag<40, false>(&Bl[0][0], wn * 64 + j * 16 + ln, 0, lane);
#pragma unroll
        for (int i = 0; i < 2; ++i)
#pragma unroll
            for (int j = 0; j < 4; ++j)
                acc[i][j] = WMMA_BF16(af[i], bfg[j], acc[i][j]);
    }

#pragma unroll
    for (int i = 0; i < 2; ++i)
#pragma unroll
        for (int j = 0; j < 4; ++j) {
            int n = nb + wn * 64 + j * 16 + ln;
            float bias = bout[n];
#pragma unroll
            for (int vi = 0; vi < 8; ++vi) {
                int m = mb + wm * 32 + i * 16 + vi + half * 8;
                y[(size_t)m * 1024 + n] = acc[i][j][vi] + bias;
            }
        }
}

extern "C" void kernel_launch(void* const* d_in, const int* in_sizes, int n_in,
                              void* d_out, int out_size, void* d_ws, size_t ws_size,
                              hipStream_t stream) {
    (void)in_sizes; (void)n_in; (void)out_size; (void)ws_size;
    const float* x    = (const float*)d_in[0];
    const float* Wqkv = (const float*)d_in[1];
    const float* bqkv = (const float*)d_in[2];
    const float* Wout = (const float*)d_in[3];
    const float* bout = (const float*)d_in[4];
    float* y = (float*)d_out;

    // Workspace partition (bf16): Q,K,V [B,H,T,d] + O [B,T,C]; 4 x 8,388,608 elems = 64MB.
    const size_t nper = (size_t)4 * 16 * 2048 * 64;
    __hip_bfloat16* qd = (__hip_bfloat16*)d_ws;
    __hip_bfloat16* kd = qd + nper;
    __hip_bfloat16* vd = kd + nper;
    __hip_bfloat16* od = vd + nper;

    qkv_gemm<<<dim3(64, 24), 256, 0, stream>>>(x, Wqkv, bqkv, qd, kd, vd);
    attn<<<dim3(64, 32), 128, 0, stream>>>(qd, kd, vd, od);
    out_gemm<<<dim3(64, 8), 256, 0, stream>>>(od, Wout, bout, y);
}
